// GATConvDense_52845277610397
// MI455X (gfx1250) — matrix operand order repaired and checked by reading, not compiled, not run on hardware
//
#include <hip/hip_runtime.h>

// ---------------------------------------------------------------------------
// GAT dense layer, fused flash-attention style.
//   - projection GEMM: V_WMMA_F32_16X16X4_F32 (full fp32, tiny: 0.4 GFLOP)
//   - PV aggregation:  V_WMMA_F32_16X16X32_F16 (9.7 GFLOP, f32 accumulate)
// N=3072 nodes, H=8 heads, D=64, D_IN=128
// ---------------------------------------------------------------------------

typedef __attribute__((ext_vector_type(2)))  float     v2f;
typedef __attribute__((ext_vector_type(8)))  float     v8f;
typedef __attribute__((ext_vector_type(16))) _Float16  v16h;

#define NN   3072
#define HH   8
#define DD   64
#define DIN  128
#define HD   (HH * DD)   // 512
#define NEGC (-1.0e9f)
#define NT   (NN / 32)   // 96 j-tiles

// ---------------------------------------------------------------------------
// Kernel 0: transpose W [512,128] -> Wt [128,512] (coalesced B gathers in proj)
// ---------------------------------------------------------------------------
__global__ void gat_transpose_w(const float* __restrict__ W, float* __restrict__ Wt) {
    int gid = blockIdx.x * blockDim.x + threadIdx.x;   // 65536
    if (gid >= HD * DIN) return;
    int o = gid >> 7;
    int k = gid & 127;
    Wt[(size_t)k * HD + o] = W[gid];
}

// ---------------------------------------------------------------------------
// Kernel 1: proj[3072,512] = x @ W^T via V_WMMA_F32_16X16X4_F32.
// 192 blocks (16 rows) x 8 waves (64 cols each), K=128 in steps of 4.
// f32 A 16x4: lane L -> M=L%16, VGPR v -> K = 2*(L/16)+v.  B mirrored.
// C/D: lane L -> N=L%16, VGPR r -> M = r + 8*(L/16).
// ---------------------------------------------------------------------------
__global__ __launch_bounds__(256) void gat_proj(const float* __restrict__ x,
                                                const float* __restrict__ Wt,
                                                float* __restrict__ proj) {
    __shared__ float sX[16][129];
    const int tid   = threadIdx.x;
    const int i0    = blockIdx.x * 16;
    const int wave  = tid >> 5;
    const int lane  = tid & 31;
    const int row16 = lane & 15;
    const int hf    = lane >> 4;

    for (int idx = tid; idx < 16 * DIN; idx += 256)
        sX[idx >> 7][idx & 127] = x[(size_t)(i0 + (idx >> 7)) * DIN + (idx & 127)];
    __syncthreads();

    v8f acc[4] = {};
    const int col0 = wave * DD;

    for (int k = 0; k < DIN; k += 4) {
        v2f A;
        A.x = sX[row16][k + hf * 2];
        A.y = sX[row16][k + hf * 2 + 1];
#pragma unroll
        for (int t = 0; t < 4; ++t) {
            const float* bp = Wt + (size_t)(k + hf * 2) * HD + col0 + t * 16 + row16;
            v2f B;
            B.x = bp[0];
            B.y = bp[HD];
            acc[t] = __builtin_amdgcn_wmma_f32_16x16x4_f32(
                false, A, false, B, (short)0, acc[t], false, false);
        }
    }

#pragma unroll
    for (int t = 0; t < 4; ++t)
#pragma unroll
        for (int r = 0; r < 8; ++r)
            proj[(size_t)(i0 + hf * 8 + r) * HD + col0 + t * 16 + row16] = acc[t][r];
}

// ---------------------------------------------------------------------------
// Kernel 2: s_src[n,h] = <proj[n,h,:], a_src[h,:]>, s_tgt likewise.
// ---------------------------------------------------------------------------
__global__ void gat_scores(const float* __restrict__ proj,
                           const float* __restrict__ a_src,
                           const float* __restrict__ a_tgt,
                           float* __restrict__ s_src, float* __restrict__ s_tgt) {
    int gid = blockIdx.x * blockDim.x + threadIdx.x;   // 24576
    if (gid >= NN * HH) return;
    int n = gid >> 3, h = gid & 7;
    const float* pp = proj + (size_t)n * HD + h * DD;
    const float* as = a_src + h * DD;
    const float* at = a_tgt + h * DD;
    float ss = 0.f, st = 0.f;
    for (int d = 0; d < DD; ++d) {
        float pv = pp[d];
        ss += pv * as[d];
        st += pv * at[d];
    }
    s_src[gid] = ss;
    s_tgt[gid] = st;
}

// ---------------------------------------------------------------------------
// Kernel 3: repack proj fp32 [N][512] -> projT f16 [512][N] (LDS tile transpose)
// so the attention kernel's B operands are contiguous 32B per lane.
// ---------------------------------------------------------------------------
__global__ __launch_bounds__(256) void gat_repack(const float* __restrict__ proj,
                                                  _Float16* __restrict__ projT) {
    __shared__ _Float16 t[64][65];
    const int nb  = blockIdx.x;   // 48 node blocks
    const int cb  = blockIdx.y;   // 8 col blocks
    const int tid = threadIdx.x;
    for (int idx = tid; idx < 64 * 64; idx += 256) {
        int nl = idx >> 6, cl = idx & 63;   // coalesced along cols
        t[cl][nl] = (_Float16)proj[(size_t)(nb * 64 + nl) * HD + cb * 64 + cl];
    }
    __syncthreads();
    for (int idx = tid; idx < 64 * 64; idx += 256) {
        int cl = idx >> 6, nl = idx & 63;   // coalesced along nodes
        projT[(size_t)(cb * 64 + cl) * NN + nb * 64 + nl] = t[cl][nl];
    }
}

// ---------------------------------------------------------------------------
// Kernel 4: fused masked attention + aggregation (flash-style online softmax),
// f16 WMMA 16x16x32 for P·V. 192 i-blocks x 8 waves; wave w = head w.
//
// A (P, 16x32 f16) layout: lane L -> M=L%16; element e -> K = 16*hf + e.
// B (V, 32x16 f16) layout: lane L -> N=L%16; element e -> K = 16*hf + e
//   -> 32 contiguous bytes of projT[col][j0+16*hf ...] per lane.
// C/D (16x16 f32):  lane L -> N=L%16; VGPR r -> M = r + 8*hf.
//
// Adjacency penalties (shared by all heads) + s_src are double-buffered in LDS
// with register-staged prefetch to overlap the HBM adj stream with compute.
// ---------------------------------------------------------------------------
__global__ __launch_bounds__(256) void gat_attn(const float* __restrict__ adj,
                                                const _Float16* __restrict__ projT,
                                                const float* __restrict__ s_src,
                                                const float* __restrict__ s_tgt,
                                                const float* __restrict__ bias,
                                                float* __restrict__ out) {
    __shared__ float sPen[2][16][33];   // (1 - adj_sl) * NEG, padded
    __shared__ float sS[2][32][HH];     // s_src tile (broadcast reads)
    __shared__ float sM[16][HH];        // running row max
    __shared__ float sL[16][HH];        // running row sum
    __shared__ float sScale[16][HH];    // per-tile rescale factor

    const int tid   = threadIdx.x;
    const int i0    = blockIdx.x * 16;
    const int head  = tid >> 5;
    const int lane  = tid & 31;
    const int row16 = lane & 15;
    const int hf    = lane >> 4;

    const float st_i = s_tgt[(size_t)(i0 + row16) * HH + head];
    if (hf == 0) { sM[row16][head] = -3.0e38f; sL[row16][head] = 0.0f; }

    // register-staged tile prefetch: 2 penalty elements + 1 s_src element/thread
    float pr[2], sr;
    auto load_tile = [&](int j0) {
#pragma unroll
        for (int q = 0; q < 2; ++q) {
            int idx = tid + q * 256;
            int ii = idx >> 5, jj = idx & 31;
            float v = adj[(size_t)(i0 + ii) * NN + j0 + jj];
            if (i0 + ii == j0 + jj) v += 1.0f;           // self loop
            v = fminf(fmaxf(v, 0.0f), 1.0f);             // clip
            pr[q] = (1.0f - v) * NEGC;                   // precomputed penalty
        }
        sr = s_src[(size_t)(j0 + (tid >> 3)) * HH + (tid & 7)];
    };
    auto store_tile = [&](int b) {
#pragma unroll
        for (int q = 0; q < 2; ++q) {
            int idx = tid + q * 256;
            sPen[b][idx >> 5][idx & 31] = pr[q];
        }
        sS[b][tid >> 3][tid & 7] = sr;
    };

    v8f acc[4] = {};

    load_tile(0);
    store_tile(0);
    load_tile(32);
    __syncthreads();

    for (int jt = 0; jt < NT; ++jt) {
        const int cur = jt & 1;

        // ---- logits in f16-A-operand order -------------------------------
        float pe[16];
        float tmax = -3.0e38f;
#pragma unroll
        for (int e = 0; e < 16; ++e) {
            int jl = e + 16 * hf;
            float z = st_i + sS[cur][jl][head];
            z = fmaxf(z, 0.2f * z);                      // leaky_relu(0.2)
            z += sPen[cur][row16][jl];                   // adjacency penalty
            pe[e] = z;
            tmax  = fmaxf(tmax, z);
        }
        tmax = fmaxf(tmax, __shfl_xor(tmax, 16, 32));    // full row max

        float m_old  = sM[row16][head];
        float m_new  = fmaxf(m_old, tmax);
        float escale = __expf(m_old - m_new);

        float rsum = 0.0f;
        v16h  Ah;
#pragma unroll
        for (int e = 0; e < 16; ++e) {
            float q = __expf(pe[e] - m_new);
            rsum += q;
            Ah[e] = (_Float16)q;
        }
        rsum += __shfl_xor(rsum, 16, 32);

        if (hf == 0) {
            sM[row16][head]     = m_new;
            sScale[row16][head] = escale;
            sL[row16][head]     = sL[row16][head] * escale + rsum;
        }
        // rescale accumulators (same-wave DS ops are in-order -> safe)
#pragma unroll
        for (int r = 0; r < 8; ++r) {
            float scl = sScale[hf * 8 + r][head];
#pragma unroll
            for (int t = 0; t < 4; ++t) acc[t][r] *= scl;
        }

        // ---- P(16x32) x V(32x64): one f16 WMMA per 16-wide n-tile --------
        const int j0 = jt * 32;
#pragma unroll
        for (int t = 0; t < 4; ++t) {
            const v16h* bp = (const v16h*)(projT +
                (size_t)(head * DD + t * 16 + row16) * NN + j0 + 16 * hf);
            acc[t] = __builtin_amdgcn_wmma_f32_16x16x32_f16(
                false, Ah, false, *bp, (short)0, acc[t], false, false);
        }

        // ---- stage next tile, prefetch tile after that -------------------
        if (jt + 1 < NT) {
            store_tile(cur ^ 1);
            if (jt + 2 < NT) load_tile((jt + 2) * 32);
        }
        __syncthreads();
    }

    // epilogue: normalize, bias, ELU, store
#pragma unroll
    for (int r = 0; r < 8; ++r) {
        int row = hf * 8 + r;
        float inv = 1.0f / sL[row][head];
#pragma unroll
        for (int t = 0; t < 4; ++t) {
            float val = acc[t][r] * inv + bias[head * DD + t * 16 + row16];
            val = val > 0.0f ? val : __expf(val) - 1.0f;   // ELU(alpha=1)
            out[(size_t)(i0 + row) * HD + head * DD + t * 16 + row16] = val;
        }
    }
}

// ---------------------------------------------------------------------------
// Launch
// ---------------------------------------------------------------------------
extern "C" void kernel_launch(void* const* d_in, const int* in_sizes, int n_in,
                              void* d_out, int out_size, void* d_ws, size_t ws_size,
                              hipStream_t stream) {
    const float* x     = (const float*)d_in[0];   // [3072,128]
    const float* adj   = (const float*)d_in[1];   // [3072,3072]
    const float* W     = (const float*)d_in[2];   // [512,128]
    const float* a_src = (const float*)d_in[3];   // [1,8,64]
    const float* a_tgt = (const float*)d_in[4];   // [1,8,64]
    const float* b     = (const float*)d_in[5];   // [512]
    float* out = (float*)d_out;                    // [3072,512]

    char* ws = (char*)d_ws;
    float*    Wt    = (float*)ws;                                      // 256 KB
    float*    proj  = (float*)(ws + 262144);                           // 6 MB
    float*    ssrc  = (float*)(ws + 262144 + 6291456);                 // 96 KB
    float*    stgt  = (float*)(ws + 262144 + 6291456 + 98304);         // 96 KB
    _Float16* projT = (_Float16*)(ws + 262144 + 6291456 + 2 * 98304);  // 3 MB

    gat_transpose_w<<<(HD * DIN + 255) / 256, 256, 0, stream>>>(W, Wt);
    gat_proj<<<NN / 16, 256, 0, stream>>>(x, Wt, proj);
    gat_scores<<<(NN * HH + 255) / 256, 256, 0, stream>>>(proj, a_src, a_tgt, ssrc, stgt);
    gat_repack<<<dim3(NN / 64, HD / 64), 256, 0, stream>>>(proj, projT);
    gat_attn<<<NN / 16, 256, 0, stream>>>(adj, projT, ssrc, stgt, b, out);
}
